// RNNAttention_87316685128117
// MI455X (gfx1250) — matrix operand order repaired and checked by reading, not compiled, not run on hardware
//
#include <hip/hip_runtime.h>
#include <hip/hip_bf16.h>

#define VOCAB  32000
#define EMBED  256
#define HIDDEN 512
#define BB     8
#define TT     512

typedef __attribute__((ext_vector_type(16))) __bf16 bf16x16;
typedef __attribute__((ext_vector_type(8)))  __bf16 bf16x8;
typedef __attribute__((ext_vector_type(8)))  float  f32x8;

// ---------------- workspace layout (bytes) ----------------
constexpr size_t OFF_WXHP = 0;                                        // 256*512 bf16
constexpr size_t OFF_WHHP = OFF_WXHP + (size_t)EMBED * HIDDEN * 2;    // 512*512 bf16
constexpr size_t OFF_FCWB = OFF_WHHP + (size_t)HIDDEN * HIDDEN * 2;   // 32000*1024 bf16
constexpr size_t OFF_X    = OFF_FCWB + (size_t)VOCAB * 2 * HIDDEN * 2;// 4096*512 f32
constexpr size_t OFF_HB   = OFF_X    + (size_t)TT * BB * HIDDEN * 4;  // 4096*512 bf16
constexpr size_t OFF_HT   = OFF_HB   + (size_t)TT * BB * HIDDEN * 2;  // 8*512*512 bf16
constexpr size_t OFF_S    = OFF_HT   + (size_t)BB * HIDDEN * TT * 2;  // 8*512*512 f32
constexpr size_t OFF_P    = OFF_S    + (size_t)BB * TT * TT * 4;      // 8*512*512 bf16
constexpr size_t OFF_CTX  = OFF_P    + (size_t)BB * TT * TT * 2;      // 8*512*512 bf16

// ---------------- helpers ----------------
__device__ __forceinline__ __bf16 f2bf(float f) {
    union { float f; unsigned u; } v; v.f = f;
    unsigned r = (v.u + 0x7FFFu + ((v.u >> 16) & 1u)) >> 16;
    unsigned short s = (unsigned short)r;
    return __builtin_bit_cast(__bf16, s);
}

__device__ __forceinline__ f32x8 wmma_bf16(bf16x16 a, bf16x16 b, f32x8 c) {
    return __builtin_amdgcn_wmma_f32_16x16x32_bf16(
        /*neg_a=*/false, a, /*neg_b=*/false, b,
        /*c_mod=*/(short)0, c, /*reuse_a=*/false, /*reuse_b=*/false);
}

// A-fragment (16x32 bf16): 16 contiguous elements at p (two 8-elem chunks at p and p+8)
__device__ __forceinline__ bf16x16 ld_a_bf(const __bf16* p) {
    bf16x8 lo = *(const bf16x8*)p;
    bf16x8 hi = *(const bf16x8*)(p + 8);
    bf16x16 a;
#pragma unroll
    for (int i = 0; i < 8; ++i) { a[i] = lo[i]; a[8 + i] = hi[i]; }
    return a;
}

// A-fragment built from fp32 source (embedding rows), converted on the fly
__device__ __forceinline__ bf16x16 ld_a_f32(const float* p) {
    const float4* q0 = (const float4*)p;
    const float4* q1 = (const float4*)(p + 8);
    float4 u0 = q0[0], u1 = q0[1], w0 = q1[0], w1 = q1[1];
    bf16x16 a;
    a[0]  = f2bf(u0.x); a[1]  = f2bf(u0.y); a[2]  = f2bf(u0.z); a[3]  = f2bf(u0.w);
    a[4]  = f2bf(u1.x); a[5]  = f2bf(u1.y); a[6]  = f2bf(u1.z); a[7]  = f2bf(u1.w);
    a[8]  = f2bf(w0.x); a[9]  = f2bf(w0.y); a[10] = f2bf(w0.z); a[11] = f2bf(w0.w);
    a[12] = f2bf(w1.x); a[13] = f2bf(w1.y); a[14] = f2bf(w1.z); a[15] = f2bf(w1.w);
    return a;
}

// B-fragment (32x16 bf16): 16 contiguous bf16 per lane (32B aligned)
__device__ __forceinline__ bf16x16 ld_b_bf(const __bf16* p) {
    return *(const bf16x16*)p;
}

// ---------------- K0a: pack row-major W[K][N] -> B-fragment-major bf16 ----------------
__global__ void pack_w_frag(const float* __restrict__ W, __bf16* __restrict__ Wp,
                            int K, int N) {
    int idx = blockIdx.x * blockDim.x + threadIdx.x;
    if (idx >= K * N) return;
    int j    = idx & 15;
    int lane = (idx >> 4) & 31;
    int tile = idx >> 9;               // tile = kt*(N/16) + nt
    int nt_n = N >> 4;
    int kt = tile / nt_n, nt = tile % nt_n;
    int n = (nt << 4) + (lane & 15);
    int k = (kt << 5) + ((lane >> 4) << 4) + j;
    Wp[idx] = f2bf(W[(size_t)k * N + n]);
}

// ---------------- K0b: fp32 -> bf16 bulk convert (fc_w) ----------------
__global__ void cvt_bf16(const float* __restrict__ src, __bf16* __restrict__ dst, int n) {
    int i = blockIdx.x * blockDim.x + threadIdx.x;
    if (i < n) dst[i] = f2bf(src[i]);
}

// ---------------- K1: X[t*8+b][n] = emb[x[b][t]] @ W_xh + b_h (embedding fused) ----------
__global__ void xgemm(const int* __restrict__ x, const float* __restrict__ emb,
                      const __bf16* __restrict__ Wxhp, const float* __restrict__ bh,
                      float* __restrict__ X) {
    int wid  = (blockIdx.x * blockDim.x + threadIdx.x) >> 5;
    int lane = threadIdx.x & 31;
    int mt = wid >> 5;                 // 0..255
    int nt = wid & 31;                 // 0..31
    int mrow = (mt << 4) + (lane & 15);
    int t = mrow >> 3, b = mrow & 7;
    int token = x[b * TT + t];
    const float* erow = emb + (size_t)token * EMBED;
    int koff  = (lane >> 4) ? 16 : 0;
    f32x8 acc = {};
#pragma unroll
    for (int kt = 0; kt < EMBED / 32; ++kt) {
        int k0 = kt << 5;
        bf16x16 a  = ld_a_f32(erow + k0 + koff);
        bf16x16 bm = ld_b_bf(Wxhp + (((size_t)(kt * (HIDDEN / 16) + nt)) << 9) + (lane << 4));
        acc = wmma_bf16(a, bm, acc);
    }
    int n = (nt << 4) + (lane & 15);
    float bias = bh[n];
    int Mb = (lane >> 4) ? 8 : 0;
#pragma unroll
    for (int r = 0; r < 8; ++r) {
        int m = (mt << 4) + Mb + r;
        X[(size_t)m * HIDDEN + n] = acc[r] + bias;
    }
}

// ---------------- K2: sequential RNN scan, one persistent workgroup ----------------
__global__ void __launch_bounds__(1024) rnn_scan(const __bf16* __restrict__ Whhp,
                                                 const float* __restrict__ X,
                                                 __bf16* __restrict__ Hb,
                                                 __bf16* __restrict__ Ht) {
    __shared__ __align__(16) __bf16 hsh[16 * HIDDEN];   // h as 16x512 A-operand (bf16)
    int tid = threadIdx.x;
    int w = tid >> 5, lane = tid & 31;
    for (int i = tid; i < 16 * HIDDEN; i += 1024) hsh[i] = f2bf(0.0f);
    __syncthreads();

    int ncol = lane & 15;
    int n    = (w << 4) + ncol;
    int koff = (lane >> 4) ? 16 : 0;
    int row  = lane & 15;

    for (int t = 0; t < TT; ++t) {
        f32x8 acc = {};
#pragma unroll
        for (int kt = 0; kt < HIDDEN / 32; ++kt) {
            int k0 = kt << 5;
            bf16x16 a  = ld_a_bf(&hsh[(size_t)row * HIDDEN + k0 + koff]);
            bf16x16 bm = ld_b_bf(Whhp + (((size_t)(kt * (HIDDEN / 16) + w)) << 9) + (lane << 4));
            acc = wmma_bf16(a, bm, acc);
        }
        __syncthreads();               // all waves done reading old h
        if (lane < 16) {
#pragma unroll
            for (int r = 0; r < 8; ++r) {       // real batch rows b = 0..7
                float v = tanhf(acc[r] + X[((size_t)t * 8 + r) * HIDDEN + n]);
                __bf16 hv = f2bf(v);
                hsh[(size_t)r * HIDDEN + n] = hv;
                Hb[((size_t)t * 8 + r) * HIDDEN + n] = hv;          // H row-major
                Ht[((size_t)r * HIDDEN + n) * TT + t] = hv;         // H transposed
            }
        } else {
#pragma unroll
            for (int r = 0; r < 8; ++r)          // pad rows 8..15 stay zero
                hsh[(size_t)(8 + r) * HIDDEN + n] = f2bf(0.0f);
        }
        __syncthreads();
    }
}

// ---------------- K3: scores S[b][t][s] = H[b,t] . H[b,s] ----------------
__global__ void attn_scores(const __bf16* __restrict__ Hb, float* __restrict__ S) {
    int wid  = (blockIdx.x * blockDim.x + threadIdx.x) >> 5;
    int lane = threadIdx.x & 31;
    int b   = wid >> 10;
    int rem = wid & 1023;
    int mt = rem >> 5, nt = rem & 31;
    int tq = (mt << 4) + (lane & 15);
    int s  = (nt << 4) + (lane & 15);
    const __bf16* arow = Hb + ((size_t)tq * 8 + b) * HIDDEN;
    const __bf16* brow = Hb + ((size_t)s  * 8 + b) * HIDDEN;   // B = H^T: contiguous K
    int koff = (lane >> 4) ? 16 : 0;
    int kbo  = (lane >> 4) ? 16 : 0;
    f32x8 acc = {};
#pragma unroll
    for (int kt = 0; kt < HIDDEN / 32; ++kt) {
        int k0 = kt << 5;
        acc = wmma_bf16(ld_a_bf(arow + k0 + koff), ld_b_bf(brow + k0 + kbo), acc);
    }
    int Mb = (lane >> 4) ? 8 : 0;
#pragma unroll
    for (int r = 0; r < 8; ++r) {
        int t2 = (mt << 4) + Mb + r;
        S[((size_t)b * TT + t2) * TT + s] = acc[r];
    }
}

// ---------------- K4: row softmax, one wave per row, P in bf16 ----------------
__global__ void softmax_rows(const float* __restrict__ S, __bf16* __restrict__ P) {
    int row  = blockIdx.x * (blockDim.x >> 5) + (threadIdx.x >> 5);
    int lane = threadIdx.x & 31;
    const float* src = S + (size_t)row * TT;
    float v[16];
    float mx = -3.0e38f;
#pragma unroll
    for (int i = 0; i < 16; ++i) { v[i] = src[lane + (i << 5)]; mx = fmaxf(mx, v[i]); }
#pragma unroll
    for (int off = 16; off; off >>= 1) mx = fmaxf(mx, __shfl_xor(mx, off, 32));
    float sum = 0.0f;
#pragma unroll
    for (int i = 0; i < 16; ++i) { v[i] = __expf(v[i] - mx); sum += v[i]; }
#pragma unroll
    for (int off = 16; off; off >>= 1) sum += __shfl_xor(sum, off, 32);
    float inv = 1.0f / sum;
    __bf16* dst = P + (size_t)row * TT;
#pragma unroll
    for (int i = 0; i < 16; ++i) dst[lane + (i << 5)] = f2bf(v[i] * inv);
}

// ---------------- K5: ctx[b][t][n] = sum_s P[b][t][s] * H[b][s][n] ----------------
__global__ void attn_ctx(const __bf16* __restrict__ P, const __bf16* __restrict__ Ht,
                         __bf16* __restrict__ Ctx) {
    int wid  = (blockIdx.x * blockDim.x + threadIdx.x) >> 5;
    int lane = threadIdx.x & 31;
    int b   = wid >> 10;
    int rem = wid & 1023;
    int mt = rem >> 5, nt = rem & 31;
    int tq = (mt << 4) + (lane & 15);
    int n  = (nt << 4) + (lane & 15);
    const __bf16* arow = P  + ((size_t)b * TT + tq) * TT;        // K = s
    const __bf16* brow = Ht + ((size_t)b * HIDDEN + n) * TT;     // Ht[b][n][s]: contiguous K
    int koff = (lane >> 4) ? 16 : 0;
    int kbo  = (lane >> 4) ? 16 : 0;
    f32x8 acc = {};
#pragma unroll
    for (int kt = 0; kt < TT / 32; ++kt) {
        int k0 = kt << 5;
        acc = wmma_bf16(ld_a_bf(arow + k0 + koff), ld_b_bf(brow + k0 + kbo), acc);
    }
    int Mb = (lane >> 4) ? 8 : 0;
#pragma unroll
    for (int r = 0; r < 8; ++r) {
        int t2 = (mt << 4) + Mb + r;
        Ctx[((size_t)b * TT + t2) * HIDDEN + n] = f2bf(acc[r]);
    }
}

// ---------------- K6: out[b,t,v] = [H|ctx] . fc_w[v] + fc_b[v]  (dominant GEMM) -------
// Register-blocked 2 M-tiles x 4 N-tiles per wave: each A fragment feeds 4 WMMAs,
// each B fragment feeds 2 WMMAs -> 8 independent WMMAs per k-step.
#define FC_MB 2
#define FC_NB 4
__global__ void fc_out(const __bf16* __restrict__ Hb, const __bf16* __restrict__ Ctx,
                       const __bf16* __restrict__ Wb, const float* __restrict__ fcb,
                       float* __restrict__ out) {
    int wid  = (blockIdx.x * blockDim.x + threadIdx.x) >> 5;
    int lane = threadIdx.x & 31;
    constexpr int NTG = VOCAB / (16 * FC_NB);          // 500 n-tile groups
    int mtg = wid / NTG;                               // 0..127 (2 m-tiles each)
    int ntg = wid % NTG;
    int koff = (lane >> 4) ? 16 : 0;
    int kbo  = (lane >> 4) ? 16 : 0;

    const __bf16* hrow[FC_MB];
    const __bf16* crow[FC_MB];
#pragma unroll
    for (int im = 0; im < FC_MB; ++im) {
        int m = ((mtg * FC_MB + im) << 4) + (lane & 15);
        int b = m >> 9, t = m & 511;
        hrow[im] = Hb  + ((size_t)t * 8 + b) * HIDDEN;
        crow[im] = Ctx + ((size_t)b * TT + t) * HIDDEN;
    }

    const __bf16* wrow[FC_NB];
    int vcol[FC_NB];
#pragma unroll
    for (int i = 0; i < FC_NB; ++i) {
        vcol[i] = ((ntg * FC_NB + i) << 4) + (lane & 15);
        wrow[i] = Wb + (size_t)vcol[i] * (2 * HIDDEN);   // fc_w row = B column: contiguous
    }

    f32x8 acc[FC_MB][FC_NB] = {};
#pragma unroll
    for (int kt = 0; kt < (2 * HIDDEN) / 32; ++kt) {
        int k0 = kt << 5;
        bf16x16 a[FC_MB];
#pragma unroll
        for (int im = 0; im < FC_MB; ++im)
            a[im] = (k0 < HIDDEN) ? ld_a_bf(hrow[im] + k0 + koff)
                                  : ld_a_bf(crow[im] + (k0 - HIDDEN) + koff);
#pragma unroll
        for (int i = 0; i < FC_NB; ++i) {
            bf16x16 bfrag = ld_b_bf(wrow[i] + k0 + kbo);
#pragma unroll
            for (int im = 0; im < FC_MB; ++im)
                acc[im][i] = wmma_bf16(a[im], bfrag, acc[im][i]);
        }
    }

    int Mb = (lane >> 4) ? 8 : 0;
#pragma unroll
    for (int im = 0; im < FC_MB; ++im) {
#pragma unroll
        for (int i = 0; i < FC_NB; ++i) {
            float bias = fcb[vcol[i]];
#pragma unroll
            for (int r = 0; r < 8; ++r) {
                int m2 = ((mtg * FC_MB + im) << 4) + Mb + r;
                out[(size_t)m2 * VOCAB + vcol[i]] = acc[im][i][r] + bias;
            }
        }
    }
}

// ---------------- host launcher ----------------
extern "C" void kernel_launch(void* const* d_in, const int* in_sizes, int n_in,
                              void* d_out, int out_size, void* d_ws, size_t ws_size,
                              hipStream_t stream) {
    (void)in_sizes; (void)n_in; (void)out_size; (void)ws_size;
    const int*   x    = (const int*)d_in[0];
    const float* emb  = (const float*)d_in[1];
    const float* Wxh  = (const float*)d_in[2];
    const float* Whh  = (const float*)d_in[3];
    const float* bh   = (const float*)d_in[4];
    const float* fcw  = (const float*)d_in[5];
    const float* fcb  = (const float*)d_in[6];
    float* out = (float*)d_out;

    char* ws = (char*)d_ws;
    __bf16* Wxhp = (__bf16*)(ws + OFF_WXHP);
    __bf16* Whhp = (__bf16*)(ws + OFF_WHHP);
    __bf16* Fcwb = (__bf16*)(ws + OFF_FCWB);
    float*  X    = (float*) (ws + OFF_X);
    __bf16* Hb   = (__bf16*)(ws + OFF_HB);
    __bf16* Ht   = (__bf16*)(ws + OFF_HT);
    float*  S    = (float*) (ws + OFF_S);
    __bf16* P    = (__bf16*)(ws + OFF_P);
    __bf16* Ctx  = (__bf16*)(ws + OFF_CTX);

    // one-shot weight packing / conversion
    pack_w_frag<<<(EMBED * HIDDEN + 255) / 256, 256, 0, stream>>>(Wxh, Wxhp, EMBED, HIDDEN);
    pack_w_frag<<<(HIDDEN * HIDDEN + 255) / 256, 256, 0, stream>>>(Whh, Whhp, HIDDEN, HIDDEN);
    cvt_bf16<<<(VOCAB * 2 * HIDDEN + 255) / 256, 256, 0, stream>>>(fcw, Fcwb, VOCAB * 2 * HIDDEN);

    // X = emb[x] @ W_xh + b_h  (hoisted out of the recurrence, embedding fused)
    xgemm<<<1024, 256, 0, stream>>>(x, emb, Wxhp, bh, X);

    // sequential scan: h_t = tanh(X_t + h @ W_hh)
    rnn_scan<<<1, 1024, 0, stream>>>(Whhp, X, Hb, Ht);

    // attention
    attn_scores<<<1024, 256, 0, stream>>>(Hb, S);
    softmax_rows<<<512, 256, 0, stream>>>(S, P);
    attn_ctx<<<1024, 256, 0, stream>>>(P, Ht, Ctx);

    // dominant FC GEMM: [4096 x 1024] @ [1024 x 32000], 2x4 tiles per wave
    constexpr int NTG = VOCAB / (16 * FC_NB);
    fc_out<<<((256 / FC_MB) * NTG) / 8, 256, 0, stream>>>(Hb, Ctx, Fcwb, fcb, out);
}
